// HeteroMusGConv_53395033423884
// MI455X (gfx1250) — matrix operand order repaired and checked by reading, not compiled, not run on hardware
//
#include <hip/hip_runtime.h>
#include <hip/hip_bf16.h>

#define DDIM 64
#define LN_EPS 1e-5f

typedef __attribute__((ext_vector_type(16))) _Float16 v16h;
typedef __attribute__((ext_vector_type(8)))  float    v8f;

union Frag16 { v16h h; uint4 q[2]; };

// A-matrix fragment (16x32 f16, M x K): lane&15 = row m.
// Lanes 0-15 hold K = k0+{0..7, 16..23}; lanes 16-31 hold K = k0+{8..15, 24..31}.
__device__ __forceinline__ v16h ldsFragA(const _Float16* p0, int ld, int row, int k0, int lane) {
    Frag16 f;
    const _Float16* p = p0 + row * ld + k0 + ((lane & 16) >> 1);
    f.q[0] = *reinterpret_cast<const uint4*>(p);
    f.q[1] = *reinterpret_cast<const uint4*>(p + 16);
    return f.h;
}

// B-matrix fragment (32x16 f16, K x N): lane&15 = col n.
// Lanes 0-15 hold K = k0+{0..7, 16..23}; lanes 16-31 hold K = k0+{8..15, 24..31}.
__device__ __forceinline__ v16h ldsFragB(const _Float16* p0, int ld, int col, int k0, int lane) {
    Frag16 f;
    const _Float16* p = p0 + col * ld + k0 + ((lane & 16) >> 1);
    f.q[0] = *reinterpret_cast<const uint4*>(p);
    f.q[1] = *reinterpret_cast<const uint4*>(p + 16);
    return f.h;
}

__device__ __forceinline__ v8f wmma16(v16h a, v16h b, v8f c) {
    return __builtin_amdgcn_wmma_f32_16x16x32_f16(false, a, false, b, (short)0, c, false, false);
}

// ---------------- Kernel 1: xl = x @ lw.T + lb ; zero h ----------------
__global__ __launch_bounds__(256) void k_lin(const float* __restrict__ x,
                                             const float* __restrict__ lw,
                                             const float* __restrict__ lb,
                                             float* __restrict__ xl,
                                             float* __restrict__ h, int nNodes) {
    __shared__ alignas(16) _Float16 wh[DDIM * DDIM];   // [n][k]
    __shared__ float lbf[DDIM];
    __shared__ alignas(16) _Float16 xh[8][16 * DDIM];
    const int tid = threadIdx.x, lane = tid & 31, wv = tid >> 5;
    for (int i = tid; i < DDIM * DDIM; i += 256) wh[i] = (_Float16)lw[i];
    if (tid < DDIM) lbf[tid] = lb[tid];
    const int base = (blockIdx.x * 8 + wv) * 16;
#pragma unroll
    for (int m = 0; m < 16; ++m) {
        int node = base + m; if (node >= nNodes) node = nNodes - 1;
        xh[wv][m * DDIM + lane]      = (_Float16)x[(size_t)node * DDIM + lane];
        xh[wv][m * DDIM + lane + 32] = (_Float16)x[(size_t)node * DDIM + lane + 32];
    }
    __syncthreads();
    const int nlo = lane & 15, hhi = lane >> 4;
    v16h a0 = ldsFragA(xh[wv], DDIM, nlo, 0, lane);
    v16h a1 = ldsFragA(xh[wv], DDIM, nlo, 32, lane);
    v8f acc[4];
#pragma unroll
    for (int t = 0; t < 4; ++t) {
        int n = nlo + 16 * t;
        v8f c;
#pragma unroll
        for (int r = 0; r < 8; ++r) c[r] = lbf[n];
        c = wmma16(a0, ldsFragB(wh, DDIM, n, 0, lane), c);
        c = wmma16(a1, ldsFragB(wh, DDIM, n, 32, lane), c);
        acc[t] = c;
    }
#pragma unroll
    for (int r = 0; r < 8; ++r) {
        int node = base + hhi * 8 + r;
        if (node < nNodes) {
#pragma unroll
            for (int t = 0; t < 4; ++t)
                xl[(size_t)node * DDIM + nlo + 16 * t] = acc[t][r];
        }
    }
#pragma unroll
    for (int m = 0; m < 16; ++m) {
        int node = base + m;
        if (node < nNodes) {
            float* hp = h + (size_t)node * 128;
            hp[lane] = 0.f; hp[lane + 32] = 0.f; hp[lane + 64] = 0.f; hp[lane + 96] = 0.f;
        }
    }
}

// ---- Kernel 2: per-edge MLP + LayerNorm + scatter-add into h ----
__global__ __launch_bounds__(256) void k_edge(const float* __restrict__ x,
                                              const float* __restrict__ xl,
                                              const float* __restrict__ w0,
                                              const float* __restrict__ b0,
                                              const float* __restrict__ g,
                                              const float* __restrict__ be,
                                              const float* __restrict__ w1,
                                              const float* __restrict__ b1,
                                              const int* __restrict__ src,
                                              const int* __restrict__ dst,
                                              float* __restrict__ h, int nEdges) {
    __shared__ alignas(16) _Float16 w0h[DDIM * DDIM];
    __shared__ alignas(16) _Float16 w1h[DDIM * DDIM];
    __shared__ float b0f[DDIM], gf[DDIM], bef[DDIM], b1f[DDIM];
    __shared__ alignas(16) _Float16 eah[8][16 * DDIM];
    __shared__ int sIdx[8][16], dIdx[8][16];
    const int tid = threadIdx.x, lane = tid & 31, wv = tid >> 5;
    for (int i = tid; i < DDIM * DDIM; i += 256) {
        w0h[i] = (_Float16)w0[i];
        w1h[i] = (_Float16)w1[i];
    }
    if (tid < DDIM) { b0f[tid] = b0[tid]; gf[tid] = g[tid]; bef[tid] = be[tid]; b1f[tid] = b1[tid]; }
    const int base = (blockIdx.x * 8 + wv) * 16;
    if (lane < 16) {
        int e = base + lane;
        int ok = e < nEdges;
        sIdx[wv][lane] = ok ? src[e] : 0;
        dIdx[wv][lane] = ok ? dst[e] : 0;
    }
    __syncthreads();
    // edge features |x_src - x_dst| -> LDS f16 (gathers hit L2: x is 25.6MB)
#pragma unroll
    for (int m = 0; m < 16; ++m) {
        const float* ps = x + (size_t)sIdx[wv][m] * DDIM;
        const float* pd = x + (size_t)dIdx[wv][m] * DDIM;
        eah[wv][m * DDIM + lane]      = (_Float16)fabsf(ps[lane] - pd[lane]);
        eah[wv][m * DDIM + lane + 32] = (_Float16)fabsf(ps[lane + 32] - pd[lane + 32]);
    }
    __syncthreads();
    const int nlo = lane & 15, hhi = lane >> 4;
    v16h a0 = ldsFragA(eah[wv], DDIM, nlo, 0, lane);
    v16h a1 = ldsFragA(eah[wv], DDIM, nlo, 32, lane);
    v8f acc[4];
#pragma unroll
    for (int t = 0; t < 4; ++t) {
        int n = nlo + 16 * t;
        v8f c;
#pragma unroll
        for (int r = 0; r < 8; ++r) c[r] = b0f[n];
        c = wmma16(a0, ldsFragB(w0h, DDIM, n, 0, lane), c);
        c = wmma16(a1, ldsFragB(w0h, DDIM, n, 32, lane), c);
#pragma unroll
        for (int r = 0; r < 8; ++r) c[r] = fmaxf(c[r], 0.f);
        acc[t] = c;
    }
    // LayerNorm over D: row m lives in one 16-lane half-group; butterfly reduce.
    float mean[8], rstd[8];
#pragma unroll
    for (int r = 0; r < 8; ++r) {
        float s = 0.f, q = 0.f;
#pragma unroll
        for (int t = 0; t < 4; ++t) { float v = acc[t][r]; s += v; q += v * v; }
        s += __shfl_xor(s, 1); q += __shfl_xor(q, 1);
        s += __shfl_xor(s, 2); q += __shfl_xor(q, 2);
        s += __shfl_xor(s, 4); q += __shfl_xor(q, 4);
        s += __shfl_xor(s, 8); q += __shfl_xor(q, 8);
        float mu  = s * (1.f / 64.f);
        float var = q * (1.f / 64.f) - mu * mu;
        mean[r] = mu;
        rstd[r] = __frsqrt_rn(fmaxf(var, 0.f) + LN_EPS);
    }
    // normalized f16 back to LDS in A layout [m][k]
#pragma unroll
    for (int t = 0; t < 4; ++t) {
        int n = nlo + 16 * t;
        float gg = gf[n], bb2 = bef[n];
#pragma unroll
        for (int r = 0; r < 8; ++r) {
            int m = hhi * 8 + r;
            eah[wv][m * DDIM + n] = (_Float16)((acc[t][r] - mean[r]) * rstd[r] * gg + bb2);
        }
    }
    __syncthreads();
    a0 = ldsFragA(eah[wv], DDIM, nlo, 0, lane);
    a1 = ldsFragA(eah[wv], DDIM, nlo, 32, lane);
#pragma unroll
    for (int t = 0; t < 4; ++t) {
        int n = nlo + 16 * t;
        v8f c;
#pragma unroll
        for (int r = 0; r < 8; ++r) c[r] = b1f[n];
        c = wmma16(a0, ldsFragB(w1h, DDIM, n, 0, lane), c);
        c = wmma16(a1, ldsFragB(w1h, DDIM, n, 32, lane), c);
        acc[t] = c;
    }
    // scatter edge result into h[dst, 64:128]  (L2-resident f32 atomics)
#pragma unroll
    for (int r = 0; r < 8; ++r) {
        int m = hhi * 8 + r;
        if (base + m < nEdges) {
            float* hp = h + (size_t)dIdx[wv][m] * 128 + 64;
#pragma unroll
            for (int t = 0; t < 4; ++t) atomicAdd(hp + nlo + 16 * t, acc[t][r]);
        }
    }
    // scatter xl[src] into h[dst, 0:64]
#pragma unroll
    for (int m = 0; m < 16; ++m) {
        if (base + m < nEdges) {
            const float* xp = xl + (size_t)sIdx[wv][m] * DDIM;
            float* hp = h + (size_t)dIdx[wv][m] * 128;
            atomicAdd(hp + lane, xp[lane]);
            atomicAdd(hp + lane + 32, xp[lane + 32]);
        }
    }
}

// ---- Kernel 3: out (+)= 0.25 * (concat(xl,h) @ pw.T + pb + bias) ----
__global__ __launch_bounds__(128) void k_proj(const float* __restrict__ xl,
                                              const float* __restrict__ h,
                                              const float* __restrict__ pw,
                                              const float* __restrict__ pb,
                                              const float* __restrict__ bb,
                                              float* __restrict__ out,
                                              int nNodes, float beta) {
    __shared__ alignas(16) _Float16 pwh[DDIM * 3 * DDIM];  // [n][k] 64 x 192
    __shared__ float pbf[DDIM];
    __shared__ alignas(16) _Float16 ain[4][16 * 3 * DDIM];
    const int tid = threadIdx.x, lane = tid & 31, wv = tid >> 5;
    for (int i = tid; i < DDIM * 3 * DDIM; i += 128) pwh[i] = (_Float16)pw[i];
    if (tid < DDIM) pbf[tid] = pb[tid] + bb[tid];
    const int base = (blockIdx.x * 4 + wv) * 16;
#pragma unroll
    for (int m = 0; m < 16; ++m) {
        int node = base + m; if (node >= nNodes) node = nNodes - 1;
        const float* xp = xl + (size_t)node * DDIM;
        const float* hp = h + (size_t)node * 128;
        _Float16* ap = ain[wv] + m * (3 * DDIM);
        ap[lane]       = (_Float16)xp[lane];
        ap[lane + 32]  = (_Float16)xp[lane + 32];
        ap[lane + 64]  = (_Float16)hp[lane];
        ap[lane + 96]  = (_Float16)hp[lane + 32];
        ap[lane + 128] = (_Float16)hp[lane + 64];
        ap[lane + 160] = (_Float16)hp[lane + 96];
    }
    __syncthreads();
    const int nlo = lane & 15, hhi = lane >> 4;
    v16h a[6];
#pragma unroll
    for (int kf = 0; kf < 6; ++kf) a[kf] = ldsFragA(ain[wv], 3 * DDIM, nlo, kf * 32, lane);
#pragma unroll
    for (int t = 0; t < 4; ++t) {
        int n = nlo + 16 * t;
        v8f c;
#pragma unroll
        for (int r = 0; r < 8; ++r) c[r] = pbf[n];
#pragma unroll
        for (int kf = 0; kf < 6; ++kf)
            c = wmma16(a[kf], ldsFragB(pwh, 3 * DDIM, n, kf * 32, lane), c);
#pragma unroll
        for (int r = 0; r < 8; ++r) {
            int node = base + hhi * 8 + r;
            if (node < nNodes) {
                float* op = out + (size_t)node * DDIM + n;
                *op = beta * (*op) + 0.25f * c[r];
            }
        }
    }
}

extern "C" void kernel_launch(void* const* d_in, const int* in_sizes, int n_in,
                              void* d_out, int out_size, void* d_ws, size_t ws_size,
                              hipStream_t stream) {
    const float* x  = (const float*)d_in[0];
    const int*   ei = (const int*)d_in[1];
    const float* lw = (const float*)d_in[2];
    const float* lb = (const float*)d_in[3];
    const float* w0 = (const float*)d_in[4];
    const float* b0 = (const float*)d_in[5];
    const float* g  = (const float*)d_in[6];
    const float* be = (const float*)d_in[7];
    const float* w1 = (const float*)d_in[8];
    const float* b1 = (const float*)d_in[9];
    const float* pw = (const float*)d_in[10];
    const float* pb = (const float*)d_in[11];
    const float* bb = (const float*)d_in[12];
    float* out = (float*)d_out;

    const int N = 100000, E = 500000;
    float* xl = (float*)d_ws;                 // [N,64]  f32  (25.6 MB)
    float* h  = xl + (size_t)N * DDIM;        // [N,128] f32  (51.2 MB)

    const int blocks1 = ((N + 15) / 16 + 7) / 8;  // 8 waves x 16 nodes / block
    const int blocks2 = ((E + 15) / 16 + 7) / 8;  // 8 waves x 16 edges / block
    const int blocks3 = ((N + 15) / 16 + 3) / 4;  // 4 waves x 16 nodes / block

    for (int t = 0; t < 4; ++t) {
        const int* src = ei + (size_t)t * 2 * E;
        const int* dst = src + E;
        k_lin<<<blocks1, 256, 0, stream>>>(x, lw + t * DDIM * DDIM, lb + t * DDIM, xl, h, N);
        k_edge<<<blocks2, 256, 0, stream>>>(x, xl,
                                            w0 + t * DDIM * DDIM, b0 + t * DDIM,
                                            g + t * DDIM, be + t * DDIM,
                                            w1 + t * DDIM * DDIM, b1 + t * DDIM,
                                            src, dst, h, E);
        k_proj<<<blocks3, 128, 0, stream>>>(xl, h, pw + (size_t)t * DDIM * 3 * DDIM,
                                            pb + t * DDIM, bb + t * DDIM, out, N,
                                            t == 0 ? 0.f : 1.f);
    }
}
